// AttentionWithEinops_19799799234652
// MI455X (gfx1250) — matrix operand order repaired and checked by reading, not compiled, not run on hardware
//
#include <hip/hip_runtime.h>
#include <hip/hip_bf16.h>

typedef _Float16 f16;
typedef __attribute__((ext_vector_type(16))) _Float16 v16h;
typedef __attribute__((ext_vector_type(8)))  _Float16 v8h;
typedef __attribute__((ext_vector_type(8)))  float    v8f;
typedef __attribute__((ext_vector_type(4)))  unsigned int u32x4;
typedef __attribute__((ext_vector_type(8)))  int      i32x8;
typedef __attribute__((ext_vector_type(4)))  int      i32x4;

#define NB   4       // batch
#define SEQ  2048    // sequence
#define DM   1024    // model dim
#define NH   16      // heads
#define DH   64      // head dim
#define HD   (NH*DH) // 1024
#define MR   (NB*SEQ)// 8192 rows
#define SCALE 0.125f // 1/sqrt(64)

static __device__ inline v8f zero8() {
    v8f z;
#pragma unroll
    for (int i = 0; i < 8; ++i) z[i] = 0.0f;
    return z;
}

static __device__ inline int rfl(unsigned int x) {
    return __builtin_amdgcn_readfirstlane((int)x);
}

// Tensor Data Mover: 2-D f16 tile (tile_d0 contiguous elems x tile_d1 rows,
// row stride stride0 elems) from global -> LDS (packed). D# per ISA ch.8.
// This toolchain exposes the 6-arg builtin (g0, g1, g2, g3, g4, cpol).
static __device__ inline void tdm_load_2d_f16(const f16* gsrc, unsigned int lds_byte,
                                              unsigned int tile_d0, unsigned int tile_d1,
                                              unsigned int stride0) {
    unsigned long long ga = (unsigned long long)(uintptr_t)gsrc;
    u32x4 g0;
    g0[0] = rfl(1u);                                   // count=1, user desc
    g0[1] = rfl(lds_byte);                             // lds_addr (bytes)
    g0[2] = rfl((unsigned int)ga);                     // global_addr[31:0]
    g0[3] = rfl((unsigned int)((ga >> 32) & 0x01FFFFFFu) | (2u << 30)); // [56:32] | type=2
    i32x8 g1;
    g1[0] = rfl(1u << 16);                             // data_size=1 (2 bytes)
    g1[1] = rfl((tile_d0 & 0xFFFFu) << 16);            // tensor_dim0[15:0]
    g1[2] = rfl((tile_d0 >> 16) | ((tile_d1 & 0xFFFFu) << 16)); // dim0 hi | dim1 lo
    g1[3] = rfl((tile_d1 >> 16) | (tile_d0 << 16));    // dim1 hi | tile_dim0
    g1[4] = rfl(tile_d1);                              // tile_dim1 | tile_dim2=0
    g1[5] = rfl(stride0);                              // tensor_dim0_stride[31:0]
    g1[6] = rfl(0u);
    g1[7] = rfl(0u);
    i32x4 zz4 = {0, 0, 0, 0};
    i32x8 zz8 = {0, 0, 0, 0, 0, 0, 0, 0};
    __builtin_amdgcn_tensor_load_to_lds(g0, g1, zz4, zz4, zz8, 0);
}

static __device__ inline unsigned int lds_off(const void* p) {
    // flat shared address: low 32 bits are the LDS offset (ISA 10.2 aperture table)
    return (unsigned int)(uintptr_t)p;
}

// A fragment (16x32 f16), contiguous k order (B uses the same):
// lanes 0-15: M=lane,   K = k0+[0..15]
// lanes16-31: M=lane-16,K = k0+[16..31]
static __device__ inline v16h load_a_frag(const f16* __restrict__ base, int lda,
                                          int m0, int k0, int lane) {
    int m    = m0 + (lane & 15);
    int koff = k0 + ((lane >> 4) << 4);
    const f16* p = base + (size_t)m * lda + koff;
    v8h lo = *(const v8h*)p;
    v8h hi = *(const v8h*)(p + 8);
    v16h r;
#pragma unroll
    for (int j = 0; j < 8; ++j) { r[j] = lo[j]; r[8 + j] = hi[j]; }
    return r;
}

// ---------------- convert / transpose kernels ----------------

__global__ void cvt_f32_to_f16(const float* __restrict__ src, f16* __restrict__ dst, int n) {
    int i = blockIdx.x * blockDim.x + threadIdx.x;
    if (i < n) dst[i] = (f16)src[i];
}

// dst[m][j] = src[j][m]  (src is [1024 j][1024 m] fp32, dst is f16)
__global__ void tcvt_f32_to_f16(const float* __restrict__ src, f16* __restrict__ dst) {
    int i = blockIdx.x * blockDim.x + threadIdx.x;
    int m = i >> 10;
    int j = i & 1023;
    dst[(size_t)m * 1024 + j] = (f16)src[(size_t)j * 1024 + m];
}

// ---------------- GEMM: C[M,N] = A[M,K] * Bt[N,K]^T ----------------
// M=8192, N=1024, K=1024. Block = 128 thr = 4 waves; block tile 128x64,
// wave tile 32x64 (2 A-frags x 4 B-frags = 8 WMMAs per 32-deep k step).
// B tile staged by the Tensor Data Mover.

template <typename OutT>
__global__ __launch_bounds__(128) void gemm_kernel(const f16* __restrict__ A,
                                                   const f16* __restrict__ Bt,
                                                   OutT* __restrict__ C) {
    const int K = 1024, N = 1024;
    __shared__ f16 Bs[64][32];

    int w    = threadIdx.x >> 5;
    int lane = threadIdx.x & 31;
    int lg   = lane >> 4, ln = lane & 15;
    int r0   = blockIdx.x * 128 + w * 32;
    int c0   = blockIdx.y * 64;

    v8f acc[2][4];
#pragma unroll
    for (int mb = 0; mb < 2; ++mb)
#pragma unroll
        for (int nb = 0; nb < 4; ++nb) acc[mb][nb] = zero8();

    for (int k0 = 0; k0 < K; k0 += 32) {
        if (w == 0) { // wave 0 drives the TDM for the 64x32 B^T tile
            tdm_load_2d_f16(Bt + (size_t)c0 * K + k0, lds_off(&Bs[0][0]),
                            /*d0=*/32, /*d1=*/64, /*stride0=*/K);
            __builtin_amdgcn_s_wait_tensorcnt(0);
        }
        if (k0 + 32 < K)
            __builtin_prefetch(A + (size_t)(r0 + (lane & 15)) * K + k0 + 32, 0, 0);
        __syncthreads();

        v16h a0 = load_a_frag(A, K, r0, k0, lane);
        v16h a1 = load_a_frag(A, K, r0 + 16, k0, lane);
#pragma unroll
        for (int nb = 0; nb < 4; ++nb) {
            v16h b = *(const v16h*)&Bs[nb * 16 + ln][lg * 16];
            acc[0][nb] = __builtin_amdgcn_wmma_f32_16x16x32_f16(
                false, a0, false, b, (short)0, acc[0][nb], false, false);
            acc[1][nb] = __builtin_amdgcn_wmma_f32_16x16x32_f16(
                false, a1, false, b, (short)0, acc[1][nb], false, false);
        }
        __syncthreads();
    }

    // C layout: VGPR i -> row + i + 8*lg, col c0 + nb*16 + ln
#pragma unroll
    for (int mb = 0; mb < 2; ++mb)
#pragma unroll
        for (int nb = 0; nb < 4; ++nb)
#pragma unroll
            for (int i = 0; i < 8; ++i) {
                int row = r0 + mb * 16 + i + 8 * lg;
                int col = c0 + nb * 16 + ln;
                C[(size_t)row * N + col] = (OutT)acc[mb][nb][i];
            }
}

// ---------------- Flash attention ----------------
// grid: (SEQ/64, NH, NB), block: 128 thr = 4 waves, wave owns 16 query rows.
// K and V tiles staged by TDM; V transposed LDS->LDS for B-fragments.

__global__ __launch_bounds__(128) void attn_kernel(const f16* __restrict__ Q,
                                                   const f16* __restrict__ Kb,
                                                   const f16* __restrict__ Vb,
                                                   f16* __restrict__ O) {
    __shared__ f16 Ks[32][64];    // [key][d]
    __shared__ f16 Vs[32][64];    // [key][d] (TDM landing zone)
    __shared__ f16 Vts[64][32];   // [d][key]
    __shared__ f16 Ps[4][16][32]; // per-wave P tile [m][k]

    int b  = blockIdx.z, h = blockIdx.y;
    int qb = blockIdx.x * 64;
    int w    = threadIdx.x >> 5;
    int lane = threadIdx.x & 31;
    int lg   = lane >> 4, ln = lane & 15;
    int r0   = qb + w * 16;

    const f16* Qbase = Q + (size_t)(b * SEQ) * HD + h * DH; // row stride HD

    v16h qa0 = load_a_frag(Qbase, HD, r0, 0, lane);
    v16h qa1 = load_a_frag(Qbase, HD, r0, 32, lane);

    v8f o[4];
#pragma unroll
    for (int nb = 0; nb < 4; ++nb) o[nb] = zero8();
    float mrun[8], lrun[8];
#pragma unroll
    for (int i = 0; i < 8; ++i) { mrun[i] = -3.0e38f; lrun[i] = 0.0f; }

    int ct = threadIdx.x >> 2;        // coop key row 0..31
    int cc = (threadIdx.x & 3) * 16;  // coop d chunk
    int ntiles = (qb + 64) / 32;

    for (int kt = 0; kt < ntiles; ++kt) {
        int kbase = kt * 32;
        if (w == 0) { // TDM: 32 keys x 64 d, row stride HD
            const f16* kp = Kb + (size_t)(b * SEQ + kbase) * HD + h * DH;
            const f16* vp = Vb + (size_t)(b * SEQ + kbase) * HD + h * DH;
            tdm_load_2d_f16(kp, lds_off(&Ks[0][0]), /*d0=*/64, /*d1=*/32, /*stride0=*/HD);
            tdm_load_2d_f16(vp, lds_off(&Vs[0][0]), /*d0=*/64, /*d1=*/32, /*stride0=*/HD);
            __builtin_amdgcn_s_wait_tensorcnt(0);
        }
        __syncthreads();
        { // cooperative transpose V tile
            v16h vv = *(const v16h*)&Vs[ct][cc];
#pragma unroll
            for (int j = 0; j < 16; ++j) Vts[cc + j][ct] = vv[j];
        }
        __syncthreads();

        // S = Q * K^T : two 16x16 score tiles (keys 0-15 and 16-31 of tile)
        v8f s0 = zero8(), s1 = zero8();
        {
            v16h b00 = *(const v16h*)&Ks[ln][lg * 16];           // sub0, d 0-31
            v16h b01 = *(const v16h*)&Ks[ln][32 + lg * 16];      // sub0, d 32-63
            v16h b10 = *(const v16h*)&Ks[16 + ln][lg * 16];      // sub1, d 0-31
            v16h b11 = *(const v16h*)&Ks[16 + ln][32 + lg * 16]; // sub1, d 32-63
            s0 = __builtin_amdgcn_wmma_f32_16x16x32_f16(false, qa0, false, b00, (short)0, s0, false, false);
            s0 = __builtin_amdgcn_wmma_f32_16x16x32_f16(false, qa1, false, b01, (short)0, s0, false, false);
            s1 = __builtin_amdgcn_wmma_f32_16x16x32_f16(false, qa0, false, b10, (short)0, s1, false, false);
            s1 = __builtin_amdgcn_wmma_f32_16x16x32_f16(false, qa1, false, b11, (short)0, s1, false, false);
        }

        // causal mask + scale, online softmax (rows split per C layout)
        float s0v[8], s1v[8];
#pragma unroll
        for (int i = 0; i < 8; ++i) {
            int row = r0 + i + 8 * lg;
            int col0 = kbase + ln;
            int col1 = col0 + 16;
            s0v[i] = (col0 <= row) ? s0[i] * SCALE : -3.0e38f;
            s1v[i] = (col1 <= row) ? s1[i] * SCALE : -3.0e38f;
        }
#pragma unroll
        for (int i = 0; i < 8; ++i) {
            float t = fmaxf(s0v[i], s1v[i]);
#pragma unroll
            for (int off = 1; off < 16; off <<= 1)
                t = fmaxf(t, __shfl_xor(t, off, 32));
            float newm = fmaxf(mrun[i], t);
            float alpha = __expf(mrun[i] - newm);
            mrun[i] = newm;
            float p0 = __expf(s0v[i] - newm);
            float p1 = __expf(s1v[i] - newm);
            float rs = p0 + p1;
#pragma unroll
            for (int off = 1; off < 16; off <<= 1)
                rs += __shfl_xor(rs, off, 32);
            lrun[i] = lrun[i] * alpha + rs;
#pragma unroll
            for (int nb = 0; nb < 4; ++nb) o[nb][i] *= alpha;
            int prow = i + 8 * lg;
            Ps[w][prow][ln]      = (f16)p0;
            Ps[w][prow][ln + 16] = (f16)p1;
        }
        // same-wave DS store->load is in-order; reload P in A-fragment layout
        v16h pa;
        {
            const f16* pp = &Ps[w][ln][lg * 16];
            v8h lo = *(const v8h*)pp;
            v8h hi = *(const v8h*)(pp + 8);
#pragma unroll
            for (int j = 0; j < 8; ++j) { pa[j] = lo[j]; pa[8 + j] = hi[j]; }
        }
        // O += P * V
#pragma unroll
        for (int nb = 0; nb < 4; ++nb) {
            v16h vb = *(const v16h*)&Vts[nb * 16 + ln][lg * 16];
            o[nb] = __builtin_amdgcn_wmma_f32_16x16x32_f16(
                false, pa, false, vb, (short)0, o[nb], false, false);
        }
        __syncthreads();
    }

    // normalize + store attn output (f16, [s][h*64+d])
    float inv[8];
#pragma unroll
    for (int i = 0; i < 8; ++i) inv[i] = (lrun[i] > 0.0f) ? (1.0f / lrun[i]) : 0.0f;
#pragma unroll
    for (int nb = 0; nb < 4; ++nb)
#pragma unroll
        for (int i = 0; i < 8; ++i) {
            int row = r0 + i + 8 * lg;
            int col = nb * 16 + ln;
            O[(size_t)(b * SEQ + row) * HD + h * DH + col] = (f16)(o[nb][i] * inv[i]);
        }
}

// ---------------- launch ----------------

extern "C" void kernel_launch(void* const* d_in, const int* in_sizes, int n_in,
                              void* d_out, int out_size, void* d_ws, size_t ws_size,
                              hipStream_t stream) {
    const float* residual = (const float*)d_in[0]; // [4,2048,1024]
    const float* W_Q = (const float*)d_in[1];      // [16,64,1024]
    const float* W_K = (const float*)d_in[2];
    const float* W_V = (const float*)d_in[3];
    const float* W_O = (const float*)d_in[4];
    float* out = (float*)d_out;                    // [4,2048,1024]

    char* ws = (char*)d_ws;
    const size_t XB = (size_t)MR * DM * sizeof(f16);   // 16 MB
    const size_t WB = (size_t)HD * DM * sizeof(f16);   // 2 MB
    f16* X   = (f16*)(ws);
    f16* Wq  = (f16*)(ws + XB);
    f16* Wk  = (f16*)(ws + XB + WB);
    f16* Wv  = (f16*)(ws + XB + 2 * WB);
    f16* WoT = (f16*)(ws + XB + 3 * WB);
    f16* Qb  = (f16*)(ws + XB + 4 * WB);
    f16* Kb  = (f16*)(ws + 2 * XB + 4 * WB);
    f16* Vb  = (f16*)(ws + 3 * XB + 4 * WB);
    f16* Ab  = (f16*)(ws + 4 * XB + 4 * WB);

    const int NX = MR * DM;      // 8388608
    const int NW = HD * DM;      // 1048576
    cvt_f32_to_f16<<<NX / 256, 256, 0, stream>>>(residual, X, NX);
    cvt_f32_to_f16<<<NW / 256, 256, 0, stream>>>(W_Q, Wq, NW);
    cvt_f32_to_f16<<<NW / 256, 256, 0, stream>>>(W_K, Wk, NW);
    cvt_f32_to_f16<<<NW / 256, 256, 0, stream>>>(W_V, Wv, NW);
    tcvt_f32_to_f16<<<NW / 256, 256, 0, stream>>>(W_O, WoT);

    dim3 ggrid(MR / 128, DM / 64); // 64 x 16
    gemm_kernel<f16><<<ggrid, 128, 0, stream>>>(X, Wq, Qb);
    gemm_kernel<f16><<<ggrid, 128, 0, stream>>>(X, Wk, Kb);
    gemm_kernel<f16><<<ggrid, 128, 0, stream>>>(X, Wv, Vb);

    dim3 agrid(SEQ / 64, NH, NB); // 32 x 16 x 4
    attn_kernel<<<agrid, 128, 0, stream>>>(Qb, Kb, Vb, Ab);

    gemm_kernel<float><<<ggrid, 128, 0, stream>>>(Ab, WoT, out);
}
